// Quantizer_31044023615834
// MI455X (gfx1250) — matrix operand order repaired and checked by reading, not compiled, not run on hardware
//
#include <hip/hip_runtime.h>

// ---------------------------------------------------------------------------
// Problem constants
//   latent [16, 512, 64, 64] f32, codebook [8,256,64], wq/wk/wv [8,64,64]
//   outputs: quantized [16,512,64,64] f32 | code [16,64,64,8] | logit [16,64,64,8,256]
// ---------------------------------------------------------------------------
#define NPIX_TOT   65536          // 16*64*64
#define TILES      4096           // NPIX_TOT / 16
#define CODE_OFF    33554432ull
#define LOGIT_OFF   34078720ull   // + 16*64*64*8

// workspace (bf16) fragment tables, element offsets
#define WQT_OFF   0u              // 8 groups * 8 frags * 512
#define KMT_OFF   32768u          // 8 * 32 * 512
#define VF_OFF    163840u         // 8 * 32 * 512
#define WS_TOTAL  294912u         // bf16 elements (576 KB)

typedef __attribute__((ext_vector_type(16))) __bf16 v16bf;
typedef __attribute__((ext_vector_type(8)))  float  v8f;
typedef __attribute__((ext_vector_type(4)))  float  v4f;
typedef __attribute__((ext_vector_type(4)))  unsigned int v4u;

union FragB {
    v16bf v;
    v4u   q[2];
    unsigned short s[16];
};

// native bf16 convert (single v_cvt op on gfx1250, vs 3-4 op integer emulation)
__device__ __forceinline__ unsigned short f2bf(float f) {
    union { __bf16 b; unsigned short u; } cv;
    cv.b = (__bf16)f;
    return cv.u;
}

__device__ __forceinline__ unsigned pcg_hash(unsigned v) {
    v = v * 747796405u + 2891336453u;
    unsigned w = ((v >> ((v >> 28) + 4u)) ^ v) * 277803737u;
    return (w >> 22) ^ w;
}

__device__ __forceinline__ float gumbelf(unsigned eid) {
    unsigned h = pcg_hash(eid ^ 0x85EBCA6Bu);
    float u = ((float)(h >> 8) + 0.5f) * (1.0f / 16777216.0f);  // (0,1)
    return -__logf(-__logf(u));
}

// ---------------------------------------------------------------------------
// Kernel 1: build bf16 B-operand tables directly in WMMA fragment order.
// B fragment (K x 16 cols): lane&15 = N column, lanes 0-15 hold K = 0..15
// (element i -> K = kc*32 + (lane>>4)*16 + i), this code's own contiguous k order; A frags use the same.
// ---------------------------------------------------------------------------
__global__ __launch_bounds__(256) void prep_frags(
    const float* __restrict__ cb, const float* __restrict__ wq,
    const float* __restrict__ wk, const float* __restrict__ wv,
    unsigned short* __restrict__ wsb)
{
    unsigned tid = blockIdx.x * 256u + threadIdx.x;
    if (tid >= WS_TOTAL) return;

    float val;
    if (tid < KMT_OFF) {
        // wqT frags: [m][nt*2+kc][lane*16+i]; K-dim = d (64), N = c (64)
        unsigned e = tid;
        int m = e >> 12;            int r = e & 4095;
        int frag = r >> 9;          int nt = frag >> 1, kc = frag & 1;
        int lane = (r >> 4) & 31;   int i = r & 15;
        int d = kc * 32 + (lane >> 4) * 16 + i;
        int c = nt * 16 + (lane & 15);
        val = wq[(m * 64 + c) * 64 + d];
    } else if (tid < VF_OFF) {
        // kmatT frags: K-dim = channel cch (64), N = code k (256)
        unsigned e = tid - KMT_OFF;
        int m = e >> 14;            int r = e & 16383;
        int frag = r >> 9;          int nt = frag >> 1, kc = frag & 1;
        int lane = (r >> 4) & 31;   int i = r & 15;
        int cch = kc * 32 + (lane >> 4) * 16 + i;
        int k   = nt * 16 + (lane & 15);
        const float* crow = cb + (m * 256 + k) * 64;
        const float* wrow = wk + (m * 64 + cch) * 64;
        float s = 0.f;
#pragma unroll
        for (int d = 0; d < 64; ++d) s = fmaf(crow[d], wrow[d], s);
        val = s;
    } else {
        // v frags: K-dim = code k (256), N = channel c (64)
        unsigned e = tid - VF_OFF;
        int m = e >> 14;            int r = e & 16383;
        int frag = r >> 9;          int nt = frag >> 3, kc = frag & 7;
        int lane = (r >> 4) & 31;   int i = r & 15;
        int k = kc * 32 + (lane >> 4) * 16 + i;
        int c = nt * 16 + (lane & 15);
        const float* crow = cb + (m * 256 + k) * 64;
        const float* wrow = wv + (m * 64 + c) * 64;
        float s = 0.f;
#pragma unroll
        for (int d = 0; d < 64; ++d) s = fmaf(crow[d], wrow[d], s);
        val = s;
    }
    wsb[tid] = f2bf(val);
}

// ---------------------------------------------------------------------------
// Kernel 2: main fused kernel. Block = 256 thr = 8 waves; wave m owns group m
// for one 16-pixel tile (contiguous along W).
// Per-wave 8 KB LDS region is time-shared: q' tile (2 KB) -> logit transpose
// buffer (8 KB, f32 [16][128]) -> e tile (8 KB, bf16 [16][256]).
// ---------------------------------------------------------------------------
__global__ __launch_bounds__(256) void quantizer_main(
    const float* __restrict__ latent,
    const unsigned short* __restrict__ wsb,
    const float* __restrict__ tptr,
    const int*   __restrict__ fptr,
    float* __restrict__ out)
{
    __shared__ __align__(16) unsigned short etile[8][4096];   // 64 KB

    const int tid  = threadIdx.x;
    const int m    = tid >> 5;
    const int lane = tid & 31;
    const int col  = lane & 15;       // N column / pixel row (A-matrix M)
    const int half = lane >> 4;

    const int pix_base = blockIdx.x << 4;
    const int n  = pix_base >> 12;
    const int hw = pix_base & 4095;
    const int h  = hw >> 6;
    const int w0 = hw & 63;           // multiple of 16

    const float invtemp = 1.0f / tptr[0];
    const int   first   = fptr[0];

    unsigned short* qt = &etile[m][0];          // [16][64] bf16 q' tile
    float*          lt = (float*)&etile[m][0];  // [16][128] f32 logit buffer
    unsigned short* et = &etile[m][0];          // [16][256] bf16 e tile

    // ---- A fragments of latent (16 px x 64 ch), direct coalesced loads ----
    FragB a0, a1;
    {
        const size_t base = ((size_t)n * 512 + (size_t)m * 64) * 4096
                          + (size_t)h * 64 + w0 + col;
#pragma unroll
        for (int i = 0; i < 16; ++i) {
            int dd = i + half * 16;   // A layout: K index
            a0.s[i] = f2bf(latent[base + (size_t)dd * 4096]);
            a1.s[i] = f2bf(latent[base + (size_t)(dd + 32) * 4096]);
        }
    }

    // ---- GEMM1: q'(16x64) = A x wqT, 8 WMMAs ----
    v8f qacc[4];
#pragma unroll
    for (int nt = 0; nt < 4; ++nt) {
        v8f c = {0.f, 0.f, 0.f, 0.f, 0.f, 0.f, 0.f, 0.f};
#pragma unroll
        for (int kc = 0; kc < 2; ++kc) {
            FragB b;
            const unsigned short* p =
                wsb + (WQT_OFF + (((m * 4 + nt) * 2 + kc) * 512u + lane * 16u));
            b.q[0] = ((const v4u*)p)[0];
            b.q[1] = ((const v4u*)p)[1];
            c = __builtin_amdgcn_wmma_f32_16x16x32_bf16(
                    false, kc ? a1.v : a0.v, false, b.v, (short)0, c, false, false);
        }
        qacc[nt] = c;
    }

    // q' D-frags -> LDS [p][c] bf16 (so A-frag re-reads are ds_load_b128)
#pragma unroll
    for (int nt = 0; nt < 4; ++nt)
#pragma unroll
        for (int j = 0; j < 8; ++j)
            qt[(j + half * 8) * 64 + nt * 16 + col] = f2bf(qacc[nt][j]);

    // ---- A fragments of q' from LDS ----
    FragB qa[2];
#pragma unroll
    for (int kc = 0; kc < 2; ++kc) {
        const unsigned short* p = qt + (col * 64 + kc * 32 + half * 16);
        qa[kc].q[0] = ((const v4u*)p)[0];        // i = 0..7
        qa[kc].q[1] = ((const v4u*)(p + 8))[0]; // i = 8..15
    }

    // ---- GEMM2: logit(16x256) = q' x kmatT, 32 WMMAs ----
    v8f acc[16];
#pragma unroll
    for (int nt = 0; nt < 16; ++nt) {
        v8f c = {0.f, 0.f, 0.f, 0.f, 0.f, 0.f, 0.f, 0.f};
#pragma unroll
        for (int kc = 0; kc < 2; ++kc) {
            FragB b;
            const unsigned short* p =
                wsb + (KMT_OFF + (((m * 16 + nt) * 2 + kc) * 512u + lane * 16u));
            b.q[0] = ((const v4u*)p)[0];
            b.q[1] = ((const v4u*)p)[1];
            c = __builtin_amdgcn_wmma_f32_16x16x32_bf16(
                    false, qa[kc].v, false, b.v, (short)0, c, false, false);
        }
        acc[nt] = c;
    }

    // ---- stream logits: transpose through LDS, 512B-contiguous b128 NT stores
    // Two 8 KB passes (k half 0..127, 128..255) reusing the per-wave region.
    {
        float* lout = out + LOGIT_OFF;
        const size_t lb0 = ((size_t)pix_base * 8 + m) * 256;
#pragma unroll
        for (int hk = 0; hk < 2; ++hk) {
#pragma unroll
            for (int nto = 0; nto < 8; ++nto) {
#pragma unroll
                for (int j = 0; j < 8; ++j)
                    lt[(j + half * 8) * 128 + nto * 16 + col] = acc[hk * 8 + nto][j];
            }
            const size_t lb = lb0 + hk * 128;
#pragma unroll
            for (int it = 0; it < 16; ++it) {   // 32 lanes cover one full row-half
                v4f val = *(const v4f*)(lt + it * 128 + lane * 4);
                __builtin_nontemporal_store(
                    val, (v4f*)(lout + lb + (size_t)it * 2048 + lane * 4));
            }
        }
    }

    // ---- code = argmax_k(raw logit), per row; halves reduce over 16 lanes ----
    {
        float* cout = out + CODE_OFF;
#pragma unroll
        for (int j = 0; j < 8; ++j) {
            float mx = acc[0][j];
            int   mk = col;
#pragma unroll
            for (int nt = 1; nt < 16; ++nt) {
                float v = acc[nt][j];
                if (v > mx) { mx = v; mk = nt * 16 + col; }
            }
#pragma unroll
            for (int s = 1; s <= 8; s <<= 1) {
                float om = __shfl_xor(mx, s, 32);
                int   ok = __shfl_xor(mk, s, 32);
                if (om > mx || (om == mx && ok < mk)) { mx = om; mk = ok; }
            }
            if (col == j)
                __builtin_nontemporal_store(
                    (float)mk, cout + (size_t)(pix_base + j + half * 8) * 8 + m);
        }
    }

    // ---- build unnormalized e-tile (bf16 in LDS) + per-row sums ----
    float rsum[8];
    const unsigned ebase = ((unsigned)pix_base * 8u + (unsigned)m) * 256u + col;
    if (first) {
        // gumbel-softmax: e = exp(s - smax), s = (logit/8 + g)/temp
#pragma unroll
        for (int nt = 0; nt < 16; ++nt)
#pragma unroll
            for (int j = 0; j < 8; ++j) {
                unsigned eid = ebase + (unsigned)(j + half * 8) * 2048u + nt * 16;
                acc[nt][j] = (acc[nt][j] * 0.125f + gumbelf(eid)) * invtemp;
            }
#pragma unroll
        for (int j = 0; j < 8; ++j) {
            float mx = acc[0][j];
#pragma unroll
            for (int nt = 1; nt < 16; ++nt) mx = fmaxf(mx, acc[nt][j]);
#pragma unroll
            for (int s = 1; s <= 8; s <<= 1) mx = fmaxf(mx, __shfl_xor(mx, s, 32));
            float rs = 0.f;
#pragma unroll
            for (int nt = 0; nt < 16; ++nt) {
                float e = __expf(acc[nt][j] - mx);
                et[(j + half * 8) * 256 + nt * 16 + col] = f2bf(e);
                rs += e;
            }
#pragma unroll
            for (int s = 1; s <= 8; s <<= 1) rs += __shfl_xor(rs, s, 32);
            rsum[j] = rs;
        }
    } else {
        // categorical sample via Gumbel-max -> one-hot e-tile
#pragma unroll
        for (int j = 0; j < 8; ++j) {
            float mx = -3.4e38f;
            int   mk = 0;
#pragma unroll
            for (int nt = 0; nt < 16; ++nt) {
                unsigned eid = ebase + (unsigned)(j + half * 8) * 2048u + nt * 16;
                float t = acc[nt][j] + gumbelf(eid);
                if (t > mx) { mx = t; mk = nt * 16 + col; }
            }
#pragma unroll
            for (int s = 1; s <= 8; s <<= 1) {
                float om = __shfl_xor(mx, s, 32);
                int   ok = __shfl_xor(mk, s, 32);
                if (om > mx || (om == mx && ok < mk)) { mx = om; mk = ok; }
            }
#pragma unroll
            for (int nt = 0; nt < 16; ++nt)
                et[(j + half * 8) * 256 + nt * 16 + col] =
                    (nt * 16 + col == mk) ? (unsigned short)0x3F80 : (unsigned short)0;
            rsum[j] = 1.0f;
        }
    }

    // ---- GEMM3: quantized(16x64) = e x v, 32 WMMAs; normalize after ----
    v8f o[4];
#pragma unroll
    for (int nt = 0; nt < 4; ++nt) {
        v8f z = {0.f, 0.f, 0.f, 0.f, 0.f, 0.f, 0.f, 0.f};
        o[nt] = z;
    }
#pragma unroll
    for (int kc = 0; kc < 8; ++kc) {
        FragB ea;
        const unsigned short* p = et + (col * 256 + kc * 32 + half * 16);
        ea.q[0] = ((const v4u*)p)[0];
        ea.q[1] = ((const v4u*)(p + 8))[0];
#pragma unroll
        for (int nt = 0; nt < 4; ++nt) {
            FragB b;
            const unsigned short* pb =
                wsb + (VF_OFF + (((m * 4 + nt) * 8 + kc) * 512u + lane * 16u));
            b.q[0] = ((const v4u*)pb)[0];
            b.q[1] = ((const v4u*)pb)[1];
            o[nt] = __builtin_amdgcn_wmma_f32_16x16x32_bf16(
                        false, ea.v, false, b.v, (short)0, o[nt], false, false);
        }
    }

    // ---- normalize rows by 1/sum(e), store quantized (b128 NT stores) ----
#pragma unroll
    for (int j = 0; j < 8; ++j) rsum[j] = 1.0f / rsum[j];
    {
        const size_t qb = ((size_t)n * 512 + (size_t)m * 64) * 4096
                        + (size_t)h * 64 + w0 + half * 8;
#pragma unroll
        for (int nt = 0; nt < 4; ++nt) {
            v4f lo = { o[nt][0] * rsum[0], o[nt][1] * rsum[1],
                       o[nt][2] * rsum[2], o[nt][3] * rsum[3] };
            v4f hi = { o[nt][4] * rsum[4], o[nt][5] * rsum[5],
                       o[nt][6] * rsum[6], o[nt][7] * rsum[7] };
            float* dst = out + qb + (size_t)(nt * 16 + col) * 4096;
            __builtin_nontemporal_store(lo, (v4f*)dst);
            __builtin_nontemporal_store(hi, (v4f*)dst + 1);
        }
    }
}

// ---------------------------------------------------------------------------
extern "C" void kernel_launch(void* const* d_in, const int* in_sizes, int n_in,
                              void* d_out, int out_size, void* d_ws, size_t ws_size,
                              hipStream_t stream)
{
    const float* latent = (const float*)d_in[0];
    const float* cb     = (const float*)d_in[1];
    const float* wq     = (const float*)d_in[2];
    const float* wk     = (const float*)d_in[3];
    const float* wv     = (const float*)d_in[4];
    const float* tptr   = (const float*)d_in[5];
    const int*   fptr   = (const int*)d_in[6];
    unsigned short* wsb = (unsigned short*)d_ws;
    float* out          = (float*)d_out;

    prep_frags<<<(WS_TOTAL + 255) / 256, 256, 0, stream>>>(cb, wq, wk, wv, wsb);
    quantizer_main<<<TILES, 256, 0, stream>>>(latent, wsb, tptr, fptr, out);
}
